// MultiheadAttention_41558103556569
// MI455X (gfx1250) — matrix operand order repaired and checked by reading, not compiled, not run on hardware
//
#include <hip/hip_runtime.h>

// ---------------------------------------------------------------------------
// MI455X (gfx1250) fused multi-head attention, bf16 WMMA + TDM path.
//
// Roofline: ~21.5 GFLOP vs ~50 MB minimal traffic @ 23.3 TB/s -> compute
// bound on the matrix units. bf16 WMMA (16x16x32) gives 8x the MACs/issue of
// fp32 WMMA (16x16x4) -> bf16 operands, fp32 accumulation. Flash-attention
// streaming softmax avoids the 268 MB score tensor. K/V tiles are staged
// into LDS once per 8-wave workgroup via the Tensor Data Mover (async,
// double buffered, TENSORcnt-synchronized) instead of per-wave VMEM loads:
// 8x less L2 traffic and copy/compute overlap.
// ---------------------------------------------------------------------------

typedef __attribute__((ext_vector_type(16))) __bf16          v16bf;
typedef __attribute__((ext_vector_type(8)))  float           v8f;
typedef __attribute__((ext_vector_type(8)))  unsigned short  us8;
typedef __attribute__((ext_vector_type(16))) unsigned short  us16;
typedef __attribute__((ext_vector_type(4)))  unsigned int    u32x4;
typedef __attribute__((ext_vector_type(8)))  int             i32x8;
typedef __attribute__((ext_vector_type(4)))  int             i32x4;

#define BB     4
#define S_LEN  2048
#define T_LEN  2048
#define CIN    256
#define HIDC   256
#define OUTC   256
#define NH     4
#define CHD    64

__device__ __forceinline__ unsigned short f2bf(float f) {
  union { float f; unsigned u; } v; v.f = f;
  unsigned r = v.u + 0x7FFFu + ((v.u >> 16) & 1u);   // round-to-nearest-even
  return (unsigned short)(r >> 16);
}

// A-fragment (16x32, row-major [M][K] source, stride lda in elements).
// Lanes 0-15 hold K {0-7,16-23} of row M=lane; lanes 16-31 hold K {8-15,24-31}.
__device__ __forceinline__ v16bf load_a_frag(const unsigned short* base, int lda) {
  int lane = threadIdx.x & 31;
  int m = lane & 15, hi = lane >> 4;
  const unsigned short* p = base + (size_t)m * lda + hi * 8;
  union { v16bf v; us8 h[2]; } u;
  u.h[0] = *(const us8*)(p);
  u.h[1] = *(const us8*)(p + 16);
  return u.v;
}

// B-fragment (32x16), source stored as [N][K] row-major (i.e. B^T), stride ldb.
// Lane pair (n, n+16) holds column n: lane n -> K {0-7,16-23}, lane n+16 -> K {8-15,24-31}.
__device__ __forceinline__ v16bf load_b_frag(const unsigned short* base, int ldb) {
  int lane = threadIdx.x & 31;
  int n = lane & 15, hi = lane >> 4;
  const unsigned short* p = base + (size_t)n * ldb + hi * 8;
  union { v16bf v; us8 h[2]; } u;
  u.h[0] = *(const us8*)(p);
  u.h[1] = *(const us8*)(p + 16);
  return u.v;
}

__device__ __forceinline__ v8f wmma_bf16(v16bf a, v16bf b, v8f c) {
  return __builtin_amdgcn_wmma_f32_16x16x32_bf16(false, a, false, b,
                                                 (short)0, c, false, false);
}

// ---------------------------------------------------------------------------
// Tensor Data Mover: async 2-D tile load Global -> LDS (bf16 elements).
// Builds the D# per CDNA5 ISA 8.3/8.4: group0 = {count, lds_addr,
// global_addr, type=2}; group1 = {data_size=2B, tensor dims, tile dims,
// dim0 stride}. Groups 2/3 (and the trailing group operand of the 6-arg
// clang-23 builtin) are zero: 2-D tensor only. Tracked by TENSORcnt.
// ---------------------------------------------------------------------------
__device__ __forceinline__ void tdm_load_2d(unsigned lds_addr,
                                            const unsigned short* gptr,
                                            unsigned tensor_d0, unsigned tensor_d1,
                                            unsigned tile_d0, unsigned tile_d1,
                                            unsigned long long stride0) {
  unsigned long long ga = (unsigned long long)(uintptr_t)gptr;
  u32x4 g0;
  g0[0] = 1u;                                             // count=1, user mode
  g0[1] = lds_addr;                                       // LDS byte address
  g0[2] = (unsigned)(ga & 0xFFFFFFFFu);                   // global_addr[31:0]
  g0[3] = (unsigned)((ga >> 32) & 0x01FFFFFFu) | 0x80000000u;  // [56:32]|type=2
  i32x8 g1;
  g1[0] = (int)(1u << 16);                                // data_size = 2 bytes
  g1[1] = (int)((tensor_d0 & 0xFFFFu) << 16);             // tensor_dim0[15:0]
  g1[2] = (int)((tensor_d0 >> 16) | ((tensor_d1 & 0xFFFFu) << 16));
  g1[3] = (int)((tensor_d1 >> 16) | (tile_d0 << 16));     // tile_dim0
  g1[4] = (int)(tile_d1 & 0xFFFFu);                       // tile_dim1 (dim2=0)
  g1[5] = (int)(stride0 & 0xFFFFFFFFull);                 // dim0_stride[31:0]
  g1[6] = (int)((stride0 >> 32) & 0xFFFFull);             // dim0_stride[47:32]
  g1[7] = 0;
  i32x4 z4 = {0, 0, 0, 0};
  i32x8 z8 = {0, 0, 0, 0, 0, 0, 0, 0};
  __builtin_amdgcn_tensor_load_to_lds(g0, g1, z4, z4, z8, 0);  // clang-23 6-arg
}

// ---------------------------------------------------------------------------
// Prep: fp32 -> bf16 conversions
// ---------------------------------------------------------------------------
__global__ void convert_transpose_kernel(const float* __restrict__ X,
                                         unsigned short* __restrict__ Xt,
                                         int C, int L, int nB) {
  // Xt[b][l][c] = bf16(X[b][c][l])
  size_t total = (size_t)nB * C * L;
  for (size_t i = (size_t)blockIdx.x * blockDim.x + threadIdx.x; i < total;
       i += (size_t)gridDim.x * blockDim.x) {
    int c = (int)(i % C);
    size_t rem = i / C;
    int l = (int)(rem % L);
    int b = (int)(rem / L);
    Xt[i] = f2bf(X[((size_t)b * C + c) * L + l]);
  }
}

__global__ void convert_flat_kernel(const float* __restrict__ X,
                                    unsigned short* __restrict__ Y, int n) {
  for (int i = blockIdx.x * blockDim.x + threadIdx.x; i < n;
       i += gridDim.x * blockDim.x)
    Y[i] = f2bf(X[i]);
}

// ---------------------------------------------------------------------------
// Projection: O[o,l] = sum_i W[o,i] X[i,l] + b[o]   (one 16x16 tile per wave)
// transposeV=0: Out[b][h][l][c]  (K/Q layout);  =1: Out[b][h][c][l]  (V layout)
// ---------------------------------------------------------------------------
__global__ __launch_bounds__(128) void proj_kernel(
    const unsigned short* __restrict__ Wbf,   // [HIDC][CIN] bf16
    const unsigned short* __restrict__ Xt,    // [B][L][CIN] bf16
    const float* __restrict__ bias,           // [HIDC]
    unsigned short* __restrict__ Out, int L, int transposeV) {
  int wave = threadIdx.x >> 5, lane = threadIdx.x & 31;
  int task = blockIdx.x * 4 + wave;
  int ntl = L / 16;
  int lt = task % ntl; task /= ntl;
  int ot = task % (HIDC / 16);
  int b  = task / (HIDC / 16);

  const unsigned short* Wt = Wbf + (size_t)ot * 16 * CIN;
  const unsigned short* Xb = Xt + ((size_t)b * L + (size_t)lt * 16) * CIN;

  v8f acc = {};
#pragma unroll
  for (int k0 = 0; k0 < CIN; k0 += 32)
    acc = wmma_bf16(load_a_frag(Wt + k0, CIN), load_b_frag(Xb + k0, CIN), acc);

  int n = lane & 15, m0 = (lane >> 4) * 8;
  int l = lt * 16 + n;
  int obase = ot * 16 + m0;
  int h = obase >> 6, cbase = obase & 63;
  if (!transposeV) {
    unsigned short* dst = Out + (((size_t)b * NH + h) * L + l) * CHD + cbase;
    us8 o;
#pragma unroll
    for (int r = 0; r < 8; r++) o[r] = f2bf(acc[r] + bias[obase + r]);
    *(us8*)dst = o;
  } else {
    unsigned short* dst =
        Out + (((size_t)b * NH + h) * CHD + cbase) * (size_t)L + l;
#pragma unroll
    for (int r = 0; r < 8; r++) dst[(size_t)r * L] = f2bf(acc[r] + bias[obase + r]);
  }
}

// ---------------------------------------------------------------------------
// Fused flash attention. One 256-thread block = (b, h, 128-wide t block);
// 8 waves each own a 16-wide t tile. Wave 0 drives the TDM: K tile [32s][64c]
// and V tile [64c][32s] are DMA'd into double-buffered LDS, all 8 waves
// consume them via ds_load WMMA fragments.
// ---------------------------------------------------------------------------
__global__ __launch_bounds__(256) void attn_kernel(
    const unsigned short* __restrict__ Khat,  // [B][H][S][CH]
    const unsigned short* __restrict__ Vhat,  // [B][H][CH][S]
    const unsigned short* __restrict__ Qhat,  // [B][H][T][CH]
    const float* __restrict__ mask,           // [B][S][T]
    unsigned short* __restrict__ Ohat) {      // [B][T][HID]
  __shared__ unsigned short kTile[2][32 * 64];    // [s][c] per buffer
  __shared__ unsigned short vTile[2][64 * 32];    // [c][s] per buffer
  __shared__ unsigned short pT_all[8][16 * 32];   // per-wave P^T tile [t][s]
  const float NEG_INF = -__builtin_inff();

  int wave = threadIdx.x >> 5, lane = threadIdx.x & 31;
  unsigned short* pT = pT_all[wave];

  int tblk = blockIdx.x & 15;            // T/128 = 16 t blocks
  int bh   = blockIdx.x >> 4;            // B*H = 16
  int b = bh >> 2, h = bh & 3;
  int t0 = tblk * 128 + wave * 16;
  int n  = lane & 15;
  int m0 = (lane >> 4) * 8;

  const unsigned short* Kbh = Khat + (size_t)bh * S_LEN * CHD;
  const unsigned short* Vbh = Vhat + (size_t)bh * CHD * S_LEN;
  const float* maskb = mask + (size_t)b * S_LEN * T_LEN;

  const unsigned short* Qbase = Qhat + ((size_t)bh * T_LEN + t0) * CHD;
  v16bf qf0 = load_b_frag(Qbase, CHD);
  v16bf qf1 = load_b_frag(Qbase + 32, CHD);

  unsigned kAddr[2] = {(unsigned)(uintptr_t)&kTile[0][0],
                       (unsigned)(uintptr_t)&kTile[1][0]};
  unsigned vAddr[2] = {(unsigned)(uintptr_t)&vTile[0][0],
                       (unsigned)(uintptr_t)&vTile[1][0]};

  // prologue: DMA first K/V tiles into buffer 0
  if (wave == 0) {
    tdm_load_2d(kAddr[0], Kbh, CHD, S_LEN, CHD, 32, CHD);
    tdm_load_2d(vAddr[0], Vbh, S_LEN, CHD, 32, CHD, S_LEN);
  }

  v8f acc0 = {}, acc1 = {}, acc2 = {}, acc3 = {};
  float m_run = NEG_INF, l_run = 0.f;
  const float scale = 0.125f;  // CH^-0.5

  for (int s0 = 0; s0 < S_LEN; s0 += 32) {
    int buf = (s0 >> 5) & 1;
    __syncthreads();  // everyone done reading buf^1 from previous iteration
    if (wave == 0) {
      if (s0 + 32 < S_LEN) {
        tdm_load_2d(kAddr[buf ^ 1], Kbh + (size_t)(s0 + 32) * CHD,
                    CHD, S_LEN, CHD, 32, CHD);
        tdm_load_2d(vAddr[buf ^ 1], Vbh + (s0 + 32),
                    S_LEN, CHD, 32, CHD, S_LEN);
        __builtin_amdgcn_s_wait_tensorcnt(2);  // current pair landed
      } else {
        __builtin_amdgcn_s_wait_tensorcnt(0);
      }
    }
    __syncthreads();  // LDS tiles for `buf` visible to all waves

    // mask[b, s, 0] for this 32-row strip (one value per lane, shfl'd below)
    float mval = maskb[(size_t)(s0 + lane) * T_LEN];
    if (s0 + 32 < S_LEN)  // -> global_prefetch_b8
      __builtin_prefetch(&maskb[(size_t)(s0 + 32 + lane) * T_LEN], 0, 1);

    const unsigned short* Kt = &kTile[buf][0];
    v16bf ka0 = load_a_frag(Kt, CHD);
    v16bf ka1 = load_a_frag(Kt + 32, CHD);
    v16bf kb0 = load_a_frag(Kt + 16 * CHD, CHD);
    v16bf kb1 = load_a_frag(Kt + 16 * CHD + 32, CHD);

    v8f sc0 = {}, sc1 = {};
    sc0 = wmma_bf16(ka0, qf0, sc0);
    sc0 = wmma_bf16(ka1, qf1, sc0);
    sc1 = wmma_bf16(kb0, qf0, sc1);
    sc1 = wmma_bf16(kb1, qf1, sc1);

    // scale + mask, per-column (t) running max
    float se[16];
    float colmax = NEG_INF;
#pragma unroll
    for (int r = 0; r < 8; r++) {
      float k0m = __shfl(mval, m0 + r, 32);
      float k1m = __shfl(mval, 16 + m0 + r, 32);
      float v0 = (k0m > 0.f) ? sc0[r] * scale : NEG_INF;
      float v1 = (k1m > 0.f) ? sc1[r] * scale : NEG_INF;
      se[r] = v0; se[8 + r] = v1;
      colmax = fmaxf(colmax, fmaxf(v0, v1));
    }
    colmax = fmaxf(colmax, __shfl_xor(colmax, 16, 32));  // lane pair shares col
    float m_new = fmaxf(m_run, colmax);
    float alpha = __expf(fminf(m_run - m_new, 0.f));     // fmin eats -inf NaNs
#pragma unroll
    for (int r = 0; r < 8; r++) {
      acc0[r] *= alpha; acc1[r] *= alpha; acc2[r] *= alpha; acc3[r] *= alpha;
    }
    l_run *= alpha;

    // probabilities -> bf16, transpose through LDS into B-fragment layout
    us8 pv0, pv1;
    float psum = 0.f;
#pragma unroll
    for (int r = 0; r < 8; r++) {
      float p0 = (se[r]     > NEG_INF) ? __expf(se[r]     - m_new) : 0.f;
      float p1 = (se[8 + r] > NEG_INF) ? __expf(se[8 + r] - m_new) : 0.f;
      psum += p0 + p1;
      pv0[r] = f2bf(p0);
      pv1[r] = f2bf(p1);
    }
    *(us8*)&pT[n * 32 + m0]      = pv0;   // P^T[t][s], s-tile 0
    *(us8*)&pT[n * 32 + 16 + m0] = pv1;   // P^T[t][s], s-tile 1
    psum += __shfl_xor(psum, 16, 32);
    l_run += psum;
    m_run = m_new;

    // same-wave LDS ops are in-order; reload as B fragment [k=s][n=t]
    v16bf pf = load_b_frag(pT, 32);

    const unsigned short* Vt = &vTile[buf][0];
    acc0 = wmma_bf16(load_a_frag(Vt,           32), pf, acc0);
    acc1 = wmma_bf16(load_a_frag(Vt + 16 * 32, 32), pf, acc1);
    acc2 = wmma_bf16(load_a_frag(Vt + 32 * 32, 32), pf, acc2);
    acc3 = wmma_bf16(load_a_frag(Vt + 48 * 32, 32), pf, acc3);
  }

  float inv_l = (l_run > 0.f) ? 1.f / l_run : 0.f;
  unsigned short* Obase =
      Ohat + ((size_t)b * T_LEN + t0 + n) * HIDC + h * CHD;
  v8f* accs[4] = {&acc0, &acc1, &acc2, &acc3};
#pragma unroll
  for (int j = 0; j < 4; j++) {
    us8 o;
#pragma unroll
    for (int r = 0; r < 8; r++) o[r] = f2bf((*accs[j])[r] * inv_l);
    *(us8*)(Obase + j * 16 + m0) = o;
  }
}

// ---------------------------------------------------------------------------
// Output projection: out[b][o][t] = (Wo . Ohat + bo) * mask[b,0,t]
// ---------------------------------------------------------------------------
__global__ __launch_bounds__(128) void outproj_kernel(
    const unsigned short* __restrict__ Wobf,  // [OUTC][HIDC] bf16
    const unsigned short* __restrict__ Ohat,  // [B][T][HIDC] bf16
    const float* __restrict__ bo,
    const float* __restrict__ mask,           // [B][S][T]
    float* __restrict__ out) {                // [B][OUTC][T]
  int wave = threadIdx.x >> 5, lane = threadIdx.x & 31;
  int task = blockIdx.x * 4 + wave;
  int lt = task % (T_LEN / 16); task /= (T_LEN / 16);
  int ot = task % (OUTC / 16);
  int b  = task / (OUTC / 16);

  const unsigned short* Wt = Wobf + (size_t)ot * 16 * HIDC;
  const unsigned short* Ob = Ohat + ((size_t)b * T_LEN + (size_t)lt * 16) * HIDC;

  v8f acc = {};
#pragma unroll
  for (int k0 = 0; k0 < HIDC; k0 += 32)
    acc = wmma_bf16(load_a_frag(Wt + k0, HIDC), load_b_frag(Ob + k0, HIDC), acc);

  int n = lane & 15, m0 = (lane >> 4) * 8;
  int t = lt * 16 + n;
  float mv = mask[(size_t)b * S_LEN * T_LEN + t];  // mask[b,0,t]
#pragma unroll
  for (int r = 0; r < 8; r++) {
    int o = ot * 16 + m0 + r;
    out[((size_t)b * OUTC + o) * T_LEN + t] = (acc[r] + bo[o]) * mv;
  }
}

// ---------------------------------------------------------------------------
extern "C" void kernel_launch(void* const* d_in, const int* in_sizes, int n_in,
                              void* d_out, int out_size, void* d_ws,
                              size_t ws_size, hipStream_t stream) {
  (void)in_sizes; (void)n_in; (void)out_size; (void)ws_size;
  const float* keys    = (const float*)d_in[0];
  const float* values  = (const float*)d_in[1];
  const float* queries = (const float*)d_in[2];
  const float* mask    = (const float*)d_in[3];
  const float* Wk = (const float*)d_in[4];  const float* bk = (const float*)d_in[5];
  const float* Wv = (const float*)d_in[6];  const float* bv = (const float*)d_in[7];
  const float* Wq = (const float*)d_in[8];  const float* bq = (const float*)d_in[9];
  const float* Wo = (const float*)d_in[10]; const float* bo = (const float*)d_in[11];
  float* out = (float*)d_out;

  char* ws = (char*)d_ws;
  size_t off = 0;
  auto salloc = [&](size_t bytes) {
    void* p = ws + off;
    off = (off + bytes + 255) & ~(size_t)255;
    return p;
  };
  const size_t XT_BYTES = (size_t)BB * S_LEN * CIN * 2;        // 4 MB each
  const size_t W_BYTES  = (size_t)HIDC * CIN * 2;              // 128 KB each
  const size_t P_BYTES  = (size_t)BB * NH * S_LEN * CHD * 2;   // 4 MB each
  unsigned short* Xkt  = (unsigned short*)salloc(XT_BYTES);
  unsigned short* Xvt  = (unsigned short*)salloc(XT_BYTES);
  unsigned short* Xqt  = (unsigned short*)salloc(XT_BYTES);
  unsigned short* WkB  = (unsigned short*)salloc(W_BYTES);
  unsigned short* WvB  = (unsigned short*)salloc(W_BYTES);
  unsigned short* WqB  = (unsigned short*)salloc(W_BYTES);
  unsigned short* WoB  = (unsigned short*)salloc(W_BYTES);
  unsigned short* Khat = (unsigned short*)salloc(P_BYTES);
  unsigned short* Vhat = (unsigned short*)salloc(P_BYTES);
  unsigned short* Qhat = (unsigned short*)salloc(P_BYTES);
  unsigned short* Ohat = (unsigned short*)salloc((size_t)BB * T_LEN * HIDC * 2);

  // 1) prep: bf16 conversion (+ transpose of activations)
  convert_transpose_kernel<<<8192, 256, 0, stream>>>(keys,    Xkt, CIN, S_LEN, BB);
  convert_transpose_kernel<<<8192, 256, 0, stream>>>(values,  Xvt, CIN, S_LEN, BB);
  convert_transpose_kernel<<<8192, 256, 0, stream>>>(queries, Xqt, CIN, T_LEN, BB);
  convert_flat_kernel<<<256, 256, 0, stream>>>(Wk, WkB, HIDC * CIN);
  convert_flat_kernel<<<256, 256, 0, stream>>>(Wv, WvB, HIDC * CIN);
  convert_flat_kernel<<<256, 256, 0, stream>>>(Wq, WqB, HIDC * CIN);
  convert_flat_kernel<<<256, 256, 0, stream>>>(Wo, WoB, OUTC * HIDC);

  // 2) projections (8192 wave-tiles each, 4 waves/block)
  proj_kernel<<<2048, 128, 0, stream>>>(WkB, Xkt, bk, Khat, S_LEN, 0);
  proj_kernel<<<2048, 128, 0, stream>>>(WvB, Xvt, bv, Vhat, S_LEN, 1);
  proj_kernel<<<2048, 128, 0, stream>>>(WqB, Xqt, bq, Qhat, T_LEN, 0);

  // 3) fused flash attention with TDM-staged K/V (256 blocks x 8 waves)
  attn_kernel<<<256, 256, 0, stream>>>(Khat, Vhat, Qhat, mask, Ohat);

  // 4) output projection + mask
  outproj_kernel<<<2048, 128, 0, stream>>>(WoB, Ohat, bo, mask, out);
}
